// HybridJLQuantizer_61151744360781
// MI455X (gfx1250) — matrix operand order repaired and checked by reading, not compiled, not run on hardware
//
#include <hip/hip_runtime.h>

// ---------------------------------------------------------------------------
// HybridJLQuantizer for MI455X (gfx1250, wave32, WMMA, async-LDS staging)
//   pass 0: keys/sketch fp32 -> bf16 (RNE) into workspace, fused row norms
//   pass 1: proj = keys @ sketch^T via v_wmma_f32_16x16x32_bf16, tiles staged
//           with GLOBAL_LOAD_ASYNC_TO_LDS_B128 (ASYNCcnt), double buffered
//   pass 2: fold per-block |max| -> scale/zero
//   pass 3: in-place 4-bit fake-quant (RNE)
// Fallback (small ws): convert-in-loop GEMM.
// ---------------------------------------------------------------------------

typedef __bf16 bf16x16 __attribute__((ext_vector_type(16)));
typedef __bf16 bf16x8  __attribute__((ext_vector_type(8)));
typedef float  f32x8   __attribute__((ext_vector_type(8)));
typedef int    i32x4   __attribute__((vector_size(16)));
typedef __attribute__((address_space(1))) i32x4 gbl_i32x4;  // global
typedef __attribute__((address_space(3))) i32x4 lds_i32x4;  // LDS

#define N_ROWS 16384
#define K_DIM  4096
#define P_DIM  1024
#define BM  128          // block tile rows (keys)
#define BPQ 128          // fast-path block tile cols
#define BPF 64           // fallback block tile cols
#define BK  32           // k-chunk per WMMA step
#define KT  (K_DIM / BK) // 128 k-steps
#define LDA 40           // halves per A row in LDS (80B, 16B aligned, padded)
#define LDB 48           // halves per B row in LDS (96B, 16B aligned, padded)

#if __has_builtin(__builtin_amdgcn_global_load_async_to_lds_b128)
#define HAVE_ASYNC 1
#else
#define HAVE_ASYNC 0
#endif

// fp32 pair -> packed bf16x2, RNE. Pure 32-bit ALU (avoids v_mov_b16 swarms).
__device__ __forceinline__ unsigned pack2(float a, float b) {
  unsigned ua = __float_as_uint(a), ub = __float_as_uint(b);
  ua += 0x7fffu + ((ua >> 16) & 1u);
  ub += 0x7fffu + ((ub >> 16) & 1u);
  return (ua >> 16) | (ub & 0xffff0000u);
}

// 16B global -> LDS copy; async (ASYNCcnt) when the gfx1250 builtin exists.
__device__ __forceinline__ void cp16_g2l(const unsigned short* g,
                                         unsigned short* l) {
#if HAVE_ASYNC
  __builtin_amdgcn_global_load_async_to_lds_b128(
      (gbl_i32x4*)g, (lds_i32x4*)l, 0, 0);
#else
  *(uint4*)l = *(const uint4*)g;
#endif
}
__device__ __forceinline__ void wait_async() {
#if HAVE_ASYNC
#if __has_builtin(__builtin_amdgcn_s_wait_asynccnt)
  __builtin_amdgcn_s_wait_asynccnt(0);
#else
  asm volatile("s_wait_asynccnt 0x0" ::: "memory");
#endif
#endif
}

// WMMA fragment gathers from LDS (ISA 7.12.2 16-bit layouts).
// A 16x32: lane L -> row M=L%16, K=(L/16)*8 + {0..7} and {16..23}
__device__ __forceinline__ bf16x16 loadFragA(const unsigned short* p) {
  bf16x16 a;
  a.lo = *(const bf16x8*)(p);
  a.hi = *(const bf16x8*)(p + 16);
  return a;
}
// B 32x16: lane L -> col N=L%16, K=(L/16)*8 + {0..7} and {16..23}
__device__ __forceinline__ bf16x16 loadFragB(const unsigned short* p) {
  bf16x16 b;
  b.lo = *(const bf16x8*)(p);
  b.hi = *(const bf16x8*)(p + 16);
  return b;
}

// ---------------------------------------------------------------------------
// Pass 0 (fast path): fp32 -> bf16 for keys & sketch, fused key row norms.
// One wave32 per row; 8 rows per block. Reads keys exactly once.
// ---------------------------------------------------------------------------
__global__ __launch_bounds__(256)
void convert_bf16_norms(const float* __restrict__ keys,
                        const float* __restrict__ sketch,
                        unsigned short* __restrict__ keysBf,
                        unsigned short* __restrict__ sketchBf,
                        float* __restrict__ norms) {
  const int wave = threadIdx.x >> 5;
  const int lane = threadIdx.x & 31;
  const int row  = blockIdx.x * 8 + wave;
  const bool isKey = row < N_ROWS;
  const int r = isKey ? row : row - N_ROWS;
  const float* src = (isKey ? keys : sketch) + (size_t)r * K_DIM;
  unsigned short* dst = (isKey ? keysBf : sketchBf) + (size_t)r * K_DIM;
  float s = 0.f;
#pragma unroll 4
  for (int i = 0; i < K_DIM / 128; ++i) {
    float4 v = *(const float4*)(src + (i * 32 + lane) * 4);
    s += v.x * v.x + v.y * v.y + v.z * v.z + v.w * v.w;
    uint2 p;
    p.x = pack2(v.x, v.y);
    p.y = pack2(v.z, v.w);
    *(uint2*)(dst + (i * 32 + lane) * 4) = p;
  }
  if (isKey) {
#pragma unroll
    for (int off = 16; off > 0; off >>= 1) s += __shfl_xor(s, off, 32);
    if (lane == 0) norms[row] = sqrtf(s);
  }
}

// ---------------------------------------------------------------------------
// Pass 1 (fast path): bf16 GEMM, async-LDS double-buffered staging.
// grid = (P/128, N/128); 8 waves; each wave: 32x64 = 2x4 WMMA tiles.
// ---------------------------------------------------------------------------
__global__ __launch_bounds__(256)
void jl_gemm_pre(const unsigned short* __restrict__ keysBf,
                 const unsigned short* __restrict__ sketchBf,
                 float* __restrict__ proj, float* __restrict__ blockMax) {
  __shared__ unsigned short As[2][BM * LDA];
  __shared__ unsigned short Bs[2][BPQ * LDB];
  __shared__ float redmax[8];

  const int tid = threadIdx.x, wave = tid >> 5, lane = tid & 31;
  const int l16 = lane & 15, lhalf = lane >> 4;
  const int blockRow = blockIdx.y * BM, blockCol = blockIdx.x * BPQ;
  const int waveM = (wave & 3) * 32, waveP = (wave >> 2) * 64;

  // A tile 128x32 bf16 = 512 x 16B chunks (2/thread); B likewise.
  auto issue = [&](int kt, int buf) {
    const unsigned short* aB = keysBf + (size_t)blockRow * K_DIM + kt * BK;
    const unsigned short* bB = sketchBf + (size_t)blockCol * K_DIM + kt * BK;
#pragma unroll
    for (int i = 0; i < 2; ++i) {
      int idx = tid + i * 256, r = idx >> 2, c = idx & 3;
      cp16_g2l(aB + (size_t)r * K_DIM + c * 8, &As[buf][r * LDA + c * 8]);
    }
#pragma unroll
    for (int i = 0; i < 2; ++i) {
      int idx = tid + i * 256, r = idx >> 2, c = idx & 3;
      cp16_g2l(bB + (size_t)r * K_DIM + c * 8, &Bs[buf][r * LDB + c * 8]);
    }
  };

  f32x8 acc[2][4] = {};
  issue(0, 0);
  wait_async();
  __syncthreads();

  for (int kt = 0; kt < KT; ++kt) {
    const int buf = kt & 1;
    if (kt + 1 < KT) issue(kt + 1, buf ^ 1);  // overlap with WMMA below

    bf16x16 aF[2], bF[4];
#pragma unroll
    for (int t = 0; t < 2; ++t)
      aF[t] = loadFragA(&As[buf][(waveM + t * 16 + l16) * LDA + lhalf * 8]);
#pragma unroll
    for (int u = 0; u < 4; ++u)
      bF[u] = loadFragB(&Bs[buf][(waveP + u * 16 + l16) * LDB + lhalf * 8]);
#pragma unroll
    for (int t = 0; t < 2; ++t)
#pragma unroll
      for (int u = 0; u < 4; ++u)
        acc[t][u] = __builtin_amdgcn_wmma_f32_16x16x32_bf16(
            false, aF[t], false, bF[u], (short)0, acc[t][u], false, false);

    if (kt + 1 < KT) wait_async();
    __syncthreads();
  }

  // Epilogue: store raw projected + block |max|.
  // C layout: VGPR j -> M = j + 8*(lane/16); N = lane%16.
  float amax = 0.f;
#pragma unroll
  for (int t = 0; t < 2; ++t)
#pragma unroll
    for (int u = 0; u < 4; ++u)
#pragma unroll
      for (int j = 0; j < 8; ++j) {
        int row = blockRow + waveM + t * 16 + j + 8 * lhalf;
        int col = blockCol + waveP + u * 16 + l16;
        float v = acc[t][u][j];
        proj[(size_t)row * P_DIM + col] = v;
        amax = fmaxf(amax, fabsf(v));
      }
#pragma unroll
  for (int off = 16; off > 0; off >>= 1)
    amax = fmaxf(amax, __shfl_xor(amax, off, 32));
  if (lane == 0) redmax[wave] = amax;
  __syncthreads();
  if (tid == 0) {
    float m = redmax[0];
#pragma unroll
    for (int w = 1; w < 8; ++w) m = fmaxf(m, redmax[w]);
    blockMax[blockIdx.y * gridDim.x + blockIdx.x] = m;
  }
}

// ---------------------------------------------------------------------------
// Fallback GEMM (ws too small): convert-in-loop, 128x64 tile.
// ---------------------------------------------------------------------------
__global__ __launch_bounds__(256)
void jl_gemm_f32(const float* __restrict__ keys,
                 const float* __restrict__ sketch,
                 float* __restrict__ proj, float* __restrict__ blockMax) {
  __shared__ unsigned short As[2][BM * LDA];
  __shared__ unsigned short Bs[2][BPF * LDB];
  __shared__ float redmax[8];

  const int tid = threadIdx.x, wave = tid >> 5, lane = tid & 31;
  const int l16 = lane & 15, lhalf = lane >> 4;
  const int blockRow = blockIdx.y * BM, blockCol = blockIdx.x * BPF;
  const int waveM = (wave & 3) * 32, waveP = (wave >> 2) * 32;

  float4 aReg[4], bReg[2];
  auto fetch = [&](int kt) {
    const float* aB = keys + (size_t)blockRow * K_DIM + kt * BK;
    const float* bB = sketch + (size_t)blockCol * K_DIM + kt * BK;
#pragma unroll
    for (int i = 0; i < 4; ++i) {
      int idx = tid + i * 256, r = idx >> 3, c4 = idx & 7;
      aReg[i] = *(const float4*)(aB + (size_t)r * K_DIM + c4 * 4);
    }
#pragma unroll
    for (int i = 0; i < 2; ++i) {
      int idx = tid + i * 256, r = idx >> 3, c4 = idx & 7;
      bReg[i] = *(const float4*)(bB + (size_t)r * K_DIM + c4 * 4);
    }
  };
  auto stage = [&](int buf) {
#pragma unroll
    for (int i = 0; i < 4; ++i) {
      int idx = tid + i * 256, r = idx >> 3, c4 = idx & 7;
      unsigned* p = (unsigned*)&As[buf][r * LDA + c4 * 4];
      p[0] = pack2(aReg[i].x, aReg[i].y);
      p[1] = pack2(aReg[i].z, aReg[i].w);
    }
#pragma unroll
    for (int i = 0; i < 2; ++i) {
      int idx = tid + i * 256, r = idx >> 3, c4 = idx & 7;
      unsigned* p = (unsigned*)&Bs[buf][r * LDB + c4 * 4];
      p[0] = pack2(bReg[i].x, bReg[i].y);
      p[1] = pack2(bReg[i].z, bReg[i].w);
    }
  };

  f32x8 acc[2][2] = {};
  fetch(0);
  stage(0);
  if (KT > 1) fetch(1);
  __syncthreads();

  for (int kt = 0; kt < KT; ++kt) {
    const int buf = kt & 1;
    bf16x16 aF[2], bF[2];
#pragma unroll
    for (int t = 0; t < 2; ++t)
      aF[t] = loadFragA(&As[buf][(waveM + t * 16 + l16) * LDA + lhalf * 8]);
#pragma unroll
    for (int u = 0; u < 2; ++u)
      bF[u] = loadFragB(&Bs[buf][(waveP + u * 16 + l16) * LDB + lhalf * 8]);
#pragma unroll
    for (int t = 0; t < 2; ++t)
#pragma unroll
      for (int u = 0; u < 2; ++u)
        acc[t][u] = __builtin_amdgcn_wmma_f32_16x16x32_bf16(
            false, aF[t], false, bF[u], (short)0, acc[t][u], false, false);
    if (kt + 1 < KT) {
      stage((kt + 1) & 1);
      if (kt + 2 < KT) fetch(kt + 2);
    }
    __syncthreads();
  }

  float amax = 0.f;
#pragma unroll
  for (int t = 0; t < 2; ++t)
#pragma unroll
    for (int u = 0; u < 2; ++u)
#pragma unroll
      for (int j = 0; j < 8; ++j) {
        int row = blockRow + waveM + t * 16 + j + 8 * lhalf;
        int col = blockCol + waveP + u * 16 + l16;
        float v = acc[t][u][j];
        proj[(size_t)row * P_DIM + col] = v;
        amax = fmaxf(amax, fabsf(v));
      }
#pragma unroll
  for (int off = 16; off > 0; off >>= 1)
    amax = fmaxf(amax, __shfl_xor(amax, off, 32));
  if (lane == 0) redmax[wave] = amax;
  __syncthreads();
  if (tid == 0) {
    float m = redmax[0];
#pragma unroll
    for (int w = 1; w < 8; ++w) m = fmaxf(m, redmax[w]);
    blockMax[blockIdx.y * gridDim.x + blockIdx.x] = m;
  }
}

// ---------------------------------------------------------------------------
// Fallback-only row norms (fast path fuses them into the convert pass).
// ---------------------------------------------------------------------------
__global__ __launch_bounds__(256)
void row_norms(const float* __restrict__ keys, float* __restrict__ norms) {
  const int wave = threadIdx.x >> 5, lane = threadIdx.x & 31;
  const int row = blockIdx.x * 8 + wave;
  const float* p = keys + (size_t)row * K_DIM;
  float s = 0.f;
#pragma unroll 4
  for (int i = 0; i < K_DIM / 128; ++i) {
    float4 v = *(const float4*)(p + (i * 32 + lane) * 4);
    s += v.x * v.x + v.y * v.y + v.z * v.z + v.w * v.w;
  }
#pragma unroll
  for (int off = 16; off > 0; off >>= 1) s += __shfl_xor(s, off, 32);
  if (lane == 0) norms[row] = sqrtf(s);
}

// ---------------------------------------------------------------------------
// Pass 2: fold per-block maxes -> scale / zero
// ---------------------------------------------------------------------------
__global__ __launch_bounds__(256)
void reduce_scale(const float* __restrict__ blockMax, int n,
                  float* __restrict__ sz, float* __restrict__ tail) {
  __shared__ float red[256];
  float m = 0.f;
  for (int i = threadIdx.x; i < n; i += 256) m = fmaxf(m, blockMax[i]);
  red[threadIdx.x] = m;
  __syncthreads();
  for (int s = 128; s > 0; s >>= 1) {
    if (threadIdx.x < s)
      red[threadIdx.x] = fmaxf(red[threadIdx.x], red[threadIdx.x + s]);
    __syncthreads();
  }
  if (threadIdx.x == 0) {
    float xmax = red[0];
    if (xmax == 0.f) xmax = 1.f;
    float scale = (2.f * xmax) / 15.f;  // (xmax - xmin)/MAXQ, xmin = -xmax
    float zero  = rintf(xmax / scale);  // RNE, matches jnp.round
    sz[0] = scale; sz[1] = zero;
    tail[0] = scale; tail[1] = zero;    // outputs 3 & 4
  }
}

// ---------------------------------------------------------------------------
// Pass 3: in-place fake-quant of projected (float4, RNE)
// ---------------------------------------------------------------------------
__global__ __launch_bounds__(256)
void fake_quant(float* __restrict__ proj, const float* __restrict__ sz) {
  const float scale = sz[0];
  const float zero  = sz[1];
  const float inv   = 1.0f / scale;
  float4* p4 = (float4*)proj;
  size_t idx = (size_t)blockIdx.x * 256 + threadIdx.x;
  float4 v = p4[idx];
  auto q1 = [&](float x) {
    float q = rintf(x * inv) + zero;
    q = fminf(fmaxf(q, 0.f), 15.f);
    return scale * (q - zero);
  };
  v.x = q1(v.x); v.y = q1(v.y); v.z = q1(v.z); v.w = q1(v.w);
  p4[idx] = v;
}

// ---------------------------------------------------------------------------
extern "C" void kernel_launch(void* const* d_in, const int* in_sizes, int n_in,
                              void* d_out, int out_size, void* d_ws,
                              size_t ws_size, hipStream_t stream) {
  const float* keys   = (const float*)d_in[0];
  const float* sketch = (const float*)d_in[1];
  float* out   = (float*)d_out;
  float* proj  = out;                           // [N, P]
  float* norms = out + (size_t)N_ROWS * P_DIM;  // [N]
  float* tail  = norms + N_ROWS;                // scale, zero

  float* ws       = (float*)d_ws;
  float* blockMax = ws;          // up to 2048 partials
  float* sz       = ws + 2048;   // [scale, zero] scratch

  const size_t bfOff   = 16384;  // byte offset of bf16 region in ws
  const size_t bfBytes = 2ull * ((size_t)N_ROWS * K_DIM + (size_t)P_DIM * K_DIM);
  const bool fast = ws_size >= bfOff + bfBytes;

  if (fast) {
    unsigned short* keysBf   = (unsigned short*)((char*)d_ws + bfOff);
    unsigned short* sketchBf = keysBf + (size_t)N_ROWS * K_DIM;
    convert_bf16_norms<<<(N_ROWS + P_DIM) / 8, 256, 0, stream>>>(
        keys, sketch, keysBf, sketchBf, norms);
    dim3 g(P_DIM / BPQ, N_ROWS / BM);  // (8, 128); x fastest -> A reuse in L2
    jl_gemm_pre<<<g, 256, 0, stream>>>(keysBf, sketchBf, proj, blockMax);
    reduce_scale<<<1, 256, 0, stream>>>(blockMax, g.x * g.y, sz, tail);
  } else {
    dim3 g(P_DIM / BPF, N_ROWS / BM);  // (16, 128)
    jl_gemm_f32<<<g, 256, 0, stream>>>(keys, sketch, proj, blockMax);
    row_norms<<<N_ROWS / 8, 256, 0, stream>>>(keys, norms);
    reduce_scale<<<1, 256, 0, stream>>>(blockMax, g.x * g.y, sz, tail);
  }
  fake_quant<<<((size_t)N_ROWS * P_DIM) / (256 * 4), 256, 0, stream>>>(proj, sz);
}
